// GraphEmbedder_72395968742042
// MI455X (gfx1250) — matrix operand order repaired and checked by reading, not compiled, not run on hardware
//
#include <hip/hip_runtime.h>
#include <hip/hip_bf16.h>
#include <stdint.h>

// ---------------------------------------------------------------------------
// DGCNN EdgeConv stack for MI455X (gfx1250, wave32, WMMA).
//   B=32, G=8192, C0=64, Cout=128, 4 layers, final max over G.
//   y = Weff@x + Wev@ev + bias(b,o);  Weff = W[:, :C] + W[:, C:2C]
//   BN(train) + LeakyReLU folded into next layer's tile load as leaky(a*y+c).
//   Intermediates in bf16 [b][g][c]; matmul via v_wmma_f32_16x16x32_bf16.
//   K and doAct are template params so A/B fragments live in fixed VGPRs
//   (no v_movrels dynamic register indexing in the WMMA loop).
// Workspace need: ~135 MB (2 x 64MB bf16 ping-pong + small tables).
// ---------------------------------------------------------------------------

typedef __attribute__((ext_vector_type(16))) __bf16     v16bf;
typedef __attribute__((ext_vector_type(8)))  float      v8f;
typedef __attribute__((ext_vector_type(4)))  unsigned   v4u;

#define EPSBN   1e-5f
#define SLOPE   0.1f
#define NB      32
#define NG      8192
#define NO      128

__device__ __forceinline__ unsigned short f2bf(float f) {
    union { float f; unsigned u; } v; v.f = f;
    unsigned u = v.u;
    u += 0x7FFFu + ((u >> 16) & 1u);          // round-to-nearest-even
    return (unsigned short)(u >> 16);
}
__device__ __forceinline__ float bf2f(unsigned short h) {
    union { unsigned u; float f; } v; v.u = ((unsigned)h) << 16;
    return v.f;
}
__device__ __forceinline__ float leaky(float x) { return x >= 0.f ? x : SLOPE * x; }

// ---------------------------------------------------------------------------
// features [b][c][G] f32  ->  act0 [b][g][64] bf16  (LDS transpose)
// ---------------------------------------------------------------------------
__global__ __launch_bounds__(256) void k_prep_act0(const float* __restrict__ F,
                                                   unsigned short* __restrict__ A0) {
    __shared__ unsigned short tile[128 * 66];
    const int b = blockIdx.y, g0 = blockIdx.x * 128, t = threadIdx.x;
    for (int i = t; i < 64 * 128; i += 256) {            // coalesced reads along g
        int c = i >> 7, g = i & 127;
        tile[g * 66 + c] = f2bf(F[((long)b * 64 + c) * NG + g0 + g]);
    }
    __syncthreads();
    unsigned* dst = (unsigned*)(A0 + ((long)b * NG + g0) * 64);
    for (int i = t; i < 128 * 32; i += 256) {            // coalesced dword writes
        int g = i >> 5, kd = i & 31;
        unsigned lo = tile[g * 66 + kd * 2], hi = tile[g * 66 + kd * 2 + 1];
        dst[(long)g * 32 + kd] = lo | (hi << 16);
    }
}

// ---------------------------------------------------------------------------
// Per-layer setup: finalize prev BN stats -> coef (a,c), zero accum,
// build Weff (bf16) and Wev (f32) for this layer.  One block, 256 threads.
// ---------------------------------------------------------------------------
__global__ __launch_bounds__(256) void k_finalize_prep(
    float* __restrict__ accum, float* __restrict__ coef,
    const float* __restrict__ gamma, const float* __restrict__ beta,
    const float* __restrict__ Wfull, unsigned short* __restrict__ Weff,
    float* __restrict__ Wev, int Cin, int haveStats, float invN) {
    const int t = threadIdx.x;
    if (t < NO) {
        float a = 1.f, c = 0.f;
        if (haveStats) {
            float m   = accum[t] * invN;
            float var = accum[NO + t] * invN - m * m;
            a = gamma[t] * rsqrtf(var + EPSBN);
            c = beta[t] - a * m;
        }
        coef[t] = a; coef[NO + t] = c;
        accum[t] = 0.f; accum[NO + t] = 0.f;             // reset for this layer
    }
    const int stride = 2 * Cin + 3;
    for (int i = t; i < NO * Cin; i += 256) {
        int o = i / Cin, cc = i % Cin;
        Weff[i] = f2bf(Wfull[o * stride + cc] + Wfull[o * stride + Cin + cc]);
    }
    for (int i = t; i < NO * 3; i += 256)
        Wev[i] = Wfull[(i / 3) * stride + 2 * Cin + (i % 3)];
}

// ---------------------------------------------------------------------------
// bias[b][o] = - sum_c W[o, Cin+c] * x0[b,c];  x0 = act(prev[b, g=0, :])
// grid = B, block = 128
// ---------------------------------------------------------------------------
__global__ __launch_bounds__(128) void k_bias(
    const float* __restrict__ Wfull, int Cin,
    const unsigned short* __restrict__ Xprev, const float* __restrict__ coef,
    int doAct, float* __restrict__ bias) {
    __shared__ float x0[128];
    const int b = blockIdx.x, t = threadIdx.x;
    if (t < Cin) {
        float v = bf2f(Xprev[(long)b * NG * Cin + t]);   // g = 0 row is contiguous
        if (doAct) v = leaky(fmaf(coef[t], v, coef[NO + t]));
        x0[t] = v;
    }
    __syncthreads();
    const int stride = 2 * Cin + 3;
    float s = 0.f;
    for (int c = 0; c < Cin; ++c) s = fmaf(Wfull[t * stride + Cin + c], x0[c], s);
    bias[b * NO + t] = -s;
}

// ---------------------------------------------------------------------------
// Main fused GEMM:  Y[b,g,o] = sum_k Weff[o,k]*act(X[b,g,k]) + Wev[o]:ev[b,:,g]
//                              + bias[b,o];  epilogue accumulates BN stats.
// 128(M) x 128(N) tile per WG, 8 waves, v_wmma_f32_16x16x32_bf16.
// K, DOACT compile-time: fragments in fixed VGPRs, fully unrolled WMMA loop.
// ---------------------------------------------------------------------------
#define SW 136   // bf16 LDS row stride (272B = 16B-aligned, conflict-free b128)

template <int K, int DOACT>
__global__ __launch_bounds__(256, 1) void k_gemm(
    const unsigned short* __restrict__ X,   // [B][G][K] bf16 (prev y or act0)
    const unsigned short* __restrict__ Wg,  // Weff [128][K] bf16
    const float* __restrict__ Wev,          // [128][3]
    const float* __restrict__ bias,         // [B][128]
    const float* __restrict__ coef,         // a[128], c[128] of previous layer
    const float* __restrict__ ev,           // [B][3][G] f32
    float* __restrict__ accum,              // sum[128], sumsq[128]
    unsigned short* __restrict__ Y) {       // [B][G][128] bf16
    __shared__ unsigned short sX[128 * SW];
    __shared__ float sEv[3 * 128];
    __shared__ float sBias[128];
    __shared__ float sWev[3 * 128];
    __shared__ float sA[128];
    __shared__ float sC[128];

    constexpr int NKT   = K / 32;           // K-tiles: 2 (layer 1) or 4
    constexpr int ROWD  = K / 2;            // dwords per staged row
    constexpr int RSH   = (K == 64) ? 5 : 6;

    const int b  = blockIdx.y;
    const int g0 = blockIdx.x * 128;
    const int t  = threadIdx.x;

    for (int i = t; i < 128; i += 256) {
        sBias[i] = bias[b * NO + i];
        sA[i] = (i < K) ? coef[i] : 1.f;
        sC[i] = (i < K) ? coef[NO + i] : 0.f;
    }
    for (int i = t; i < 3 * 128; i += 256) sWev[i] = Wev[i];
    for (int i = t; i < 3 * 128; i += 256) {
        int j = i >> 7, g = i & 127;
        sEv[j * 128 + g] = ev[((long)b * 3 + j) * NG + g0 + g];
    }
    __syncthreads();                                      // sA/sC ready

    // ---- stage X tile into LDS with fused BN-affine + LeakyReLU ----
    {
        const unsigned* src = (const unsigned*)(X + ((long)b * NG + g0) * K);
        unsigned* sXd = (unsigned*)sX;
        #pragma unroll
        for (int ii = 0; ii < 128 * ROWD / 256; ++ii) {
            const int i = ii * 256 + t;                   // tile is contiguous
            const int g = i >> RSH, kd = i & (ROWD - 1);
            unsigned u = src[i];
            if (DOACT) {
                const int k = kd * 2;
                float lo = leaky(fmaf(sA[k],     bf2f((unsigned short)(u & 0xFFFF)), sC[k]));
                float hi = leaky(fmaf(sA[k + 1], bf2f((unsigned short)(u >> 16)),    sC[k + 1]));
                u = (unsigned)f2bf(lo) | ((unsigned)f2bf(hi) << 16);
            }
            sXd[g * (SW / 2) + kd] = u;
        }
    }
    __syncthreads();

    const int wave  = t >> 5;
    const int lane  = t & 31;
    const int Ms    = wave * 16;       // this wave's 16-channel M strip
    const int lhalf = lane >> 4;
    const int l16   = lane & 15;

    // ---- A fragments (Weff rows) straight from L2-resident global ----
    v16bf afrag[NKT];
    {
        const v4u* wrow = (const v4u*)(Wg + (long)(Ms + l16) * K);
        #pragma unroll
        for (int kt = 0; kt < NKT; ++kt) {
            union { v4u q[2]; v16bf v; } f;
            const int q = kt * 4 + lhalf;                 // ISA 16-bit A layout
            f.q[0] = wrow[q];
            f.q[1] = wrow[q + 2];
            afrag[kt] = f.v;
        }
    }

    v8f acc[8];
    #pragma unroll
    for (int i = 0; i < 8; ++i) acc[i] = (v8f){0.f,0.f,0.f,0.f,0.f,0.f,0.f,0.f};

    #pragma unroll
    for (int nt = 0; nt < 8; ++nt) {
        const v4u* xrow = (const v4u*)(sX + (nt * 16 + l16) * SW);
        #pragma unroll
        for (int kt = 0; kt < NKT; ++kt) {
            union { v4u q[2]; v16bf v; } f;               // ISA 16-bit B layout
            const int q = kt * 4 + lhalf;
            f.q[0] = xrow[q];
            f.q[1] = xrow[q + 2];
            acc[nt] = __builtin_amdgcn_wmma_f32_16x16x32_bf16(
                false, afrag[kt], false, f.v, (short)0, acc[nt], false, false);
        }
    }

    // ---- epilogue: ev term + bias, BN-stat partials, bf16 store ----
    float psum[8], psq[8];
    #pragma unroll
    for (int r = 0; r < 8; ++r) { psum[r] = 0.f; psq[r] = 0.f; }
    const int obase = Ms + lhalf * 8;                     // 8 consecutive channels
    #pragma unroll
    for (int nt = 0; nt < 8; ++nt) {
        const int gcol = nt * 16 + l16;
        const float e0 = sEv[gcol], e1 = sEv[128 + gcol], e2 = sEv[256 + gcol];
        unsigned outp[4];
        #pragma unroll
        for (int r = 0; r < 8; ++r) {
            const int o = obase + r;
            float v = acc[nt][r] + sBias[o];
            v = fmaf(sWev[o * 3 + 0], e0, v);
            v = fmaf(sWev[o * 3 + 1], e1, v);
            v = fmaf(sWev[o * 3 + 2], e2, v);
            psum[r] += v;
            psq[r]   = fmaf(v, v, psq[r]);
            unsigned short h = f2bf(v);
            if (r & 1) outp[r >> 1] |= ((unsigned)h) << 16; else outp[r >> 1] = h;
        }
        v4u ov = { outp[0], outp[1], outp[2], outp[3] };
        *(v4u*)(Y + ((long)b * NG + g0 + gcol) * NO + obase) = ov;   // b128 store
    }
    // reduce across the 16-lane half (same channel, different g), then atomics
    #pragma unroll
    for (int r = 0; r < 8; ++r) {
        float s = psum[r], q = psq[r];
        #pragma unroll
        for (int m = 1; m < 16; m <<= 1) {
            s += __shfl_xor(s, m, 32);
            q += __shfl_xor(q, m, 32);
        }
        if (l16 == 0) {
            const int o = obase + r;
            atomicAdd(&accum[o], s);
            atomicAdd(&accum[NO + o], q);
        }
    }
}

// ---------------------------------------------------------------------------
// Final max over G with fused BN+LeakyReLU (two-stage, deterministic)
// ---------------------------------------------------------------------------
__global__ __launch_bounds__(256) void k_max_part(
    const unsigned short* __restrict__ Y, const float* __restrict__ coef,
    float* __restrict__ part) {
    const int b = blockIdx.y, chunk = blockIdx.x, t = threadIdx.x;
    const int o = t & 127, sub = t >> 7;                  // 2 threads per channel
    const float a = coef[o], c = coef[NO + o];
    float m = -3.4e38f;
    const int gbeg = chunk * 256 + sub * 128;
    const unsigned short* p = Y + ((long)b * NG + gbeg) * NO + o;
    for (int g = 0; g < 128; ++g)
        m = fmaxf(m, leaky(fmaf(a, bf2f(p[(long)g * NO]), c)));
    __shared__ float red[256];
    red[t] = m;
    __syncthreads();
    if (t < 128) part[((long)b * 32 + chunk) * NO + o] = fmaxf(red[t], red[t + 128]);
}

__global__ __launch_bounds__(128) void k_max_final(const float* __restrict__ part,
                                                   float* __restrict__ out) {
    const int b = blockIdx.x, o = threadIdx.x;
    float m = -3.4e38f;
    for (int ch = 0; ch < 32; ++ch)
        m = fmaxf(m, part[((long)b * 32 + ch) * NO + o]);
    out[b * NO + o] = m;
}

// ---------------------------------------------------------------------------
extern "C" void kernel_launch(void* const* d_in, const int* in_sizes, int n_in,
                              void* d_out, int out_size, void* d_ws, size_t ws_size,
                              hipStream_t stream) {
    const float* F  = (const float*)d_in[0];
    const float* ev = (const float*)d_in[1];
    const float* W[4]  = { (const float*)d_in[2], (const float*)d_in[3],
                           (const float*)d_in[4], (const float*)d_in[5] };
    const float* gm[4] = { (const float*)d_in[6],  (const float*)d_in[8],
                           (const float*)d_in[10], (const float*)d_in[12] };
    const float* bt[4] = { (const float*)d_in[7],  (const float*)d_in[9],
                           (const float*)d_in[11], (const float*)d_in[13] };

    char* ws = (char*)d_ws;
    const size_t BUF = (size_t)NB * NG * NO * sizeof(unsigned short);   // 64 MB
    unsigned short* bufA = (unsigned short*)ws;
    unsigned short* bufB = (unsigned short*)(ws + BUF);
    size_t off = 2 * BUF;
    unsigned short* weff = (unsigned short*)(ws + off); off += 128 * 128 * 2;
    float* wev   = (float*)(ws + off); off = (off + 1536 + 255) & ~(size_t)255;
    float* bias  = (float*)(ws + off); off += NB * NO * 4;
    float* accum = (float*)(ws + off); off += 2 * NO * 4;
    float* coef  = (float*)(ws + off); off += 2 * NO * 4;
    float* part  = (float*)(ws + off);

    const float invN = 1.0f / (float)(NB * NG);
    const dim3 gGemm(NG / 128, NB);

    // stage 0: features -> bf16 [b][g][64]
    k_prep_act0<<<dim3(NG / 128, NB), 256, 0, stream>>>(F, bufA);

    // layer 1: K=64, raw input (no BN/act on load)
    k_finalize_prep<<<1, 256, 0, stream>>>(accum, coef, gm[0], bt[0],
                                           W[0], weff, wev, 64, 0, invN);
    k_bias<<<NB, 128, 0, stream>>>(W[0], 64, bufA, coef, 0, bias);
    k_gemm<64, 0><<<gGemm, 256, 0, stream>>>(bufA, weff, wev, bias, coef, ev,
                                             accum, bufB);
    // layers 2..4: K=128, fused BN+leaky on load
    unsigned short* src = bufB;
    unsigned short* dst = bufA;
    for (int l = 1; l < 4; ++l) {
        k_finalize_prep<<<1, 256, 0, stream>>>(accum, coef, gm[l - 1], bt[l - 1],
                                               W[l], weff, wev, 128, 1, invN);
        k_bias<<<NB, 128, 0, stream>>>(W[l], 128, src, coef, 1, bias);
        k_gemm<128, 1><<<gGemm, 256, 0, stream>>>(src, weff, wev, bias, coef, ev,
                                                  accum, dst);
        unsigned short* tmp = src; src = dst; dst = tmp;
    }
    // finalize layer-4 BN coef, then fused max over G
    k_finalize_prep<<<1, 256, 0, stream>>>(accum, coef, gm[3], bt[3],
                                           W[3], weff, wev, NO, 1, invN);
    k_max_part<<<dim3(NG / 256, NB), 256, 0, stream>>>(src, coef, part);
    k_max_final<<<NB, 128, 0, stream>>>(part, (float*)d_out);
}
